// DK_Conv2D_33079838113953
// MI455X (gfx1250) — matrix operand order repaired and checked by reading, not compiled, not run on hardware
//
#include <hip/hip_runtime.h>
#include <hip/hip_bf16.h>

typedef _Float16 v8h  __attribute__((ext_vector_type(8)));
typedef _Float16 v16h __attribute__((ext_vector_type(16)));
typedef float    v8f  __attribute__((ext_vector_type(8)));

#define BATCH 8
#define NCH   64
#define MCH   64
#define HDIM  128
#define WDIM  128
#define KTAPS 9          // 3x3
#define KRED  576        // 64 ch * 9 taps
#define BSTR  592        // 576 + 16 halves pad -> 1184B pixel stride (32B aligned)

// ---------------------------------------------------------------------------
// Repack weight (m, n, 3, 3) fp32 -> f16 [m][k'] with k' = p*64 + c
// ---------------------------------------------------------------------------
__global__ __launch_bounds__(256) void wpack_kernel(const float* __restrict__ w,
                                                    _Float16* __restrict__ wp) {
    int idx = blockIdx.x * 256 + threadIdx.x;          // 64*576 = 36864
    if (idx >= MCH * KRED) return;
    int m = idx / KRED, r = idx % KRED;
    int p = r >> 6, c = r & 63;
    wp[idx] = (_Float16)w[(m * NCH + c) * KTAPS + p];
}

// ---------------------------------------------------------------------------
// Dynamic bias: global avg pool + 2x (1x1 conv 64->64), relu between.
// One block per batch.
// ---------------------------------------------------------------------------
__global__ __launch_bounds__(256) void bias_kernel(const float* __restrict__ x,
                                                   const float* __restrict__ w1,
                                                   const float* __restrict__ b1,
                                                   const float* __restrict__ w2,
                                                   const float* __restrict__ b2,
                                                   float* __restrict__ biasw) {
    __shared__ float part[256];
    __shared__ float g[64];
    __shared__ float h[64];
    const int b = blockIdx.x, tid = threadIdx.x;
    const int n = tid >> 2, sub = tid & 3;
    const float* xp = x + ((size_t)(b * NCH + n)) * (HDIM * WDIM);
    float s = 0.f;
    for (int i = sub; i < HDIM * WDIM; i += 4) s += xp[i];
    part[tid] = s;
    __syncthreads();
    if (sub == 0)
        g[n] = (part[tid] + part[tid + 1] + part[tid + 2] + part[tid + 3]) *
               (1.f / (HDIM * WDIM));
    __syncthreads();
    if (tid < 64) {
        float a = b1[tid];
        for (int u = 0; u < 64; ++u) a += w1[tid * 64 + u] * g[u];
        h[tid] = fmaxf(a, 0.f);
    }
    __syncthreads();
    if (tid < 64) {
        float a = b2[tid];
        for (int u = 0; u < 64; ++u) a += w2[tid * 64 + u] * h[u];
        biasw[b * 64 + tid] = a;
    }
}

// ---------------------------------------------------------------------------
// Attention: atw = sigmoid(1x1(relu(1x1(relu(conv3x3(x)))))) -> (b,9,H,W)
// One block = one image row (128 threads), weights staged in LDS.
// ---------------------------------------------------------------------------
__global__ __launch_bounds__(128) void attn_kernel(const float* __restrict__ x,
                                                   const float* __restrict__ w1,
                                                   const float* __restrict__ b1,
                                                   const float* __restrict__ w2,
                                                   const float* __restrict__ b2,
                                                   const float* __restrict__ w3,
                                                   const float* __restrict__ b3,
                                                   float* __restrict__ atw) {
    __shared__ float w1s[KTAPS * NCH * KTAPS];   // 5184
    __shared__ float w2s[81], w3s[81];
    __shared__ float b1s[9], b2s[9], b3s[9];
    const int tid = threadIdx.x;                 // col 0..127
    const int row = blockIdx.x & (HDIM - 1);
    const int b   = blockIdx.x >> 7;
    for (int i = tid; i < KTAPS * NCH * KTAPS; i += 128) w1s[i] = w1[i];
    for (int i = tid; i < 81; i += 128) { w2s[i] = w2[i]; w3s[i] = w3[i]; }
    if (tid < 9) { b1s[tid] = b1[tid]; b2s[tid] = b2[tid]; b3s[tid] = b3[tid]; }
    __syncthreads();

    float acc[9];
#pragma unroll
    for (int t = 0; t < 9; ++t) acc[t] = b1s[t];
    for (int c = 0; c < NCH; ++c) {
        const float* xc = x + ((size_t)(b * NCH + c)) * (HDIM * WDIM);
        float v[9];
#pragma unroll
        for (int ki = 0; ki < 3; ++ki) {
            int yy = row + ki - 1;
#pragma unroll
            for (int kj = 0; kj < 3; ++kj) {
                int xx = tid + kj - 1;
                v[ki * 3 + kj] =
                    ((unsigned)yy < (unsigned)HDIM && (unsigned)xx < (unsigned)WDIM)
                        ? xc[yy * WDIM + xx] : 0.f;
            }
        }
#pragma unroll
        for (int t = 0; t < 9; ++t) {
            const float* wr = &w1s[(t * NCH + c) * 9];
#pragma unroll
            for (int u = 0; u < 9; ++u) acc[t] = fmaf(v[u], wr[u], acc[t]);
        }
    }
    float a2[9];
#pragma unroll
    for (int t = 0; t < 9; ++t) acc[t] = fmaxf(acc[t], 0.f);
#pragma unroll
    for (int t = 0; t < 9; ++t) {
        float a = b2s[t];
#pragma unroll
        for (int u = 0; u < 9; ++u) a = fmaf(w2s[t * 9 + u], acc[u], a);
        a2[t] = fmaxf(a, 0.f);
    }
#pragma unroll
    for (int t = 0; t < 9; ++t) {
        float a = b3s[t];
#pragma unroll
        for (int u = 0; u < 9; ++u) a = fmaf(w3s[t * 9 + u], a2[u], a);
        float sg = 1.f / (1.f + __expf(-a));
        atw[(((size_t)(b * KTAPS + t)) * HDIM + row) * WDIM + tid] = sg;
    }
}

// ---------------------------------------------------------------------------
// Main: attention-scaled im2col GEMM via V_WMMA_F32_16X16X32_F16.
// Workgroup: 256 threads = 8 waves; output tile 64 m x 32 pixels of one row.
// K reduction order k' = p*64 + c so each 32-chunk has one attention tap.
// ---------------------------------------------------------------------------
__global__ __launch_bounds__(256) void main_wmma_kernel(
    const float* __restrict__ x, const _Float16* __restrict__ wp,
    const float* __restrict__ atw, const float* __restrict__ biasw,
    float* __restrict__ out) {
    __shared__ __align__(32) _Float16 Bt[32 * BSTR];   // 37888 B
    __shared__ float atw_s[9 * 32];
    __shared__ float bias_s[64];

    const int tid = threadIdx.x;
    const int wg  = blockIdx.x;
    const int tx  = wg & 3;                 // 4 pixel-tiles per row
    const int y   = (wg >> 2) & (HDIM - 1);
    const int b   = wg >> 9;
    const int x0  = tx * 32;

    if (tid < 9 * 32) {
        int p = tid >> 5, pix = tid & 31;
        atw_s[tid] = atw[(((size_t)(b * KTAPS + p)) * HDIM + y) * WDIM + x0 + pix];
    }
    if (tid < 64) bias_s[tid] = biasw[b * 64 + tid];
    __syncthreads();

    // Build B tile: Bt[pix][k'] = x[b][c][y+ki-1][x0+pix+kj-1] * atw[p][pix], f16
    {
        const int pix = tid & 31;
        for (int kp = tid >> 5; kp < KRED; kp += 8) {
            int p = kp >> 6, c = kp & 63;
            int ki = p / 3, kj = p % 3;
            int yy = y + ki - 1;
            int xx = x0 + pix + kj - 1;
            float v = 0.f;
            if ((unsigned)yy < (unsigned)HDIM && (unsigned)xx < (unsigned)WDIM)
                v = x[(((size_t)(b * NCH + c)) * HDIM + yy) * WDIM + xx];
            Bt[pix * BSTR + kp] = (_Float16)(v * atw_s[p * 32 + pix]);
        }
    }
    __syncthreads();

    // 8 waves: wave = (m_tile 0..3, n_sub 0..1) -> one 16x16 C tile, K=576
    const int lane  = tid & 31;
    const int wv    = tid >> 5;
    const int mt    = wv & 3;
    const int ns    = wv >> 2;
    const int lm    = lane & 15;
    const int lhalf = lane >> 4;

    const _Float16* wrow = wp + (size_t)(mt * 16 + lm) * KRED;
    const _Float16* brow = Bt + (ns * 16 + lm) * BSTR;

    v8f acc = {};
#pragma unroll 6
    for (int kc = 0; kc < KRED / 32; ++kc) {
        // A fragment: row = lane%16, e -> K = 32kc + 16*lhalf + e (contiguous 32B)
        v16h a  = *(const v16h*)(wrow + kc * 32 + lhalf * 16);
        // B fragment: col = lane%16, e -> K = 32kc + 16*lhalf + e (contiguous 32B)
        v16h bf = *(const v16h*)(brow + kc * 32 + lhalf * 16);
        acc = __builtin_amdgcn_wmma_f32_16x16x32_f16(
            false, a, false, bf, (short)0, acc, false, false);
    }

    // C layout: lane -> pixel col = lane%16; VGPR r -> M = mt*16 + r + 8*lhalf
    const int pixg = x0 + ns * 16 + lm;
#pragma unroll
    for (int r = 0; r < 8; ++r) {
        int m = mt * 16 + r + 8 * lhalf;
        float v = acc[r] + bias_s[m];
        out[(((size_t)(b * MCH + m)) * HDIM + y) * WDIM + pixg] = v;
    }
}

// ---------------------------------------------------------------------------
extern "C" void kernel_launch(void* const* d_in, const int* in_sizes, int n_in,
                              void* d_out, int out_size, void* d_ws, size_t ws_size,
                              hipStream_t stream) {
    const float* x    = (const float*)d_in[0];
    const float* a1w1 = (const float*)d_in[1];
    const float* a1b1 = (const float*)d_in[2];
    const float* a1w2 = (const float*)d_in[3];
    const float* a1b2 = (const float*)d_in[4];
    const float* a1w3 = (const float*)d_in[5];
    const float* a1b3 = (const float*)d_in[6];
    const float* a3w1 = (const float*)d_in[7];
    const float* a3b1 = (const float*)d_in[8];
    const float* a3w2 = (const float*)d_in[9];
    const float* a3b2 = (const float*)d_in[10];
    const float* wgt  = (const float*)d_in[11];
    float* out = (float*)d_out;

    char* ws = (char*)d_ws;
    _Float16* wpack = (_Float16*)ws;                     // 73728 B
    float*    biasw = (float*)(ws + 73728);              // 2048 B
    float*    atw   = (float*)(ws + 75776);              // 8*9*128*128*4 B

    wpack_kernel<<<(MCH * KRED + 255) / 256, 256, 0, stream>>>(wgt, wpack);
    bias_kernel<<<BATCH, 256, 0, stream>>>(x, a3w1, a3b1, a3w2, a3b2, biasw);
    attn_kernel<<<BATCH * HDIM, 128, 0, stream>>>(x, a1w1, a1b1, a1w2, a1b2,
                                                  a1w3, a1b3, atw);
    main_wmma_kernel<<<BATCH * HDIM * 4, 256, 0, stream>>>(x, wpack, atw, biasw, out);
}
